// MultiheadAttention_82446192214279
// MI455X (gfx1250) — compile-verified, matrix operand order checked by reading, not run on hardware
//
#include <hip/hip_runtime.h>
#include <hip/hip_bf16.h>
#include <math.h>

// ---------------------------------------------------------------------------
// CDNA5 (gfx1250) bf16-WMMA implementation of gated GQA attention w/ RoPE,
// ALiBi and the reference's (faithful) inverted sliding-window mask.
// Wave32; v_wmma_f32_16x16x32_bf16 for all matmul work; Tensor Data Mover
// (tensor_load_to_lds + s_wait_tensorcnt) for GEMM A-operand tiles.
// ---------------------------------------------------------------------------

typedef __bf16 bf16;
typedef __attribute__((ext_vector_type(16))) __bf16 v16bf;
typedef __attribute__((ext_vector_type(8)))  __bf16 v8bf;
typedef __attribute__((ext_vector_type(8)))  float  v8f;
typedef __attribute__((ext_vector_type(4)))  unsigned int u32x4;
typedef __attribute__((ext_vector_type(4)))  int i32x4;
typedef __attribute__((ext_vector_type(8)))  int i32x8;

#define DEV_INLINE __device__ __forceinline__

#if defined(__gfx1250__) && __has_builtin(__builtin_amdgcn_tensor_load_to_lds)
#define USE_TDM 1
#else
#define USE_TDM 0
#endif

constexpr int   Bc    = 2;
constexpr int   Sc    = 2048;
constexpr int   DIMc  = 2048;
constexpr int   Hc    = 16;
constexpr int   KVHc  = 8;
constexpr int   HDc   = 128;
constexpr float SCALEc = 0.08838834764831845f;   // 1/sqrt(128)
constexpr float NEGc   = -1.0e9f;

// Build a 16x32 bf16 WMMA A/B fragment from an LDS row (ISA 7.12.2 layout):
// lane m = lane&15, kbase = (lane>=16)?8:0 ; vec elems 0..7 -> k=kbase+0..7,
// elems 8..15 -> k=kbase+16..23.  Caller passes &row[kk + kbase].
DEV_INLINE v16bf make_frag(const bf16* base) {
  v8bf lo = *(const v8bf*)(base);
  v8bf hi = *(const v8bf*)(base + 16);
  v16bf r;
#pragma unroll
  for (int i = 0; i < 8; ++i) { r[i] = lo[i]; r[i + 8] = hi[i]; }
  return r;
}

// ---------------------------------------------------------------------------
// Kernel 1: complexity gate -> bf16.  xg[row] = bf16(x[row]*sigmoid(sum|x|))
// ---------------------------------------------------------------------------
__global__ __launch_bounds__(256)
void gate_kernel(const float* __restrict__ x, bf16* __restrict__ xg) {
  const int row = blockIdx.x;
  const int tid = threadIdx.x;
  const float* xr = x + (size_t)row * DIMc;
  __shared__ float red[256];
  float s = 0.f;
  for (int i = tid; i < DIMc; i += 256) s += fabsf(xr[i]);
  red[tid] = s;
  __syncthreads();
#pragma unroll
  for (int st = 128; st > 0; st >>= 1) {
    if (tid < st) red[tid] += red[tid + st];
    __syncthreads();
  }
  const float g = 1.0f / (1.0f + __expf(-red[0]));
  bf16* o = xg + (size_t)row * DIMc;
  for (int i = tid; i < DIMc; i += 256) o[i] = (bf16)(xr[i] * g);
}

// ---------------------------------------------------------------------------
// Kernel 1b: plain f32 -> bf16 copy (for the ungated activations)
// ---------------------------------------------------------------------------
__global__ __launch_bounds__(256)
void f32_to_bf16_kernel(const float* __restrict__ src, bf16* __restrict__ dst,
                        long long n8) {
  const long long i = (long long)blockIdx.x * blockDim.x + threadIdx.x;
  if (i >= n8) return;
  const float4* s = (const float4*)(src + i * 8);
  float4 f0 = s[0], f1 = s[1];
  v8bf o;
  o[0] = (bf16)f0.x; o[1] = (bf16)f0.y; o[2] = (bf16)f0.z; o[3] = (bf16)f0.w;
  o[4] = (bf16)f1.x; o[5] = (bf16)f1.y; o[6] = (bf16)f1.z; o[7] = (bf16)f1.w;
  *(v8bf*)(dst + i * 8) = o;
}

// ---------------------------------------------------------------------------
// Kernel 2: GEMM  C[MxN](f32) = A[MxK](bf16) * B[KxN](f32->bf16) (+bias).
// Block tile 128x128, K-step 32, 8 wave32 waves, 32x64 sub-tile per wave.
// A-operand tile is DMA'd by the Tensor Data Mover with LDS row padding.
// ---------------------------------------------------------------------------
constexpr int GLDA = 40;   // LDS row stride (elems): 32 data + 8 pad, 16B aligned

__global__ __launch_bounds__(256)
void gemm_bf16_wmma_kernel(const bf16* __restrict__ A, const float* __restrict__ Bm,
                           const float* __restrict__ bias, float* __restrict__ C,
                           int M, int N, int K) {
  __shared__ bf16 lds_a[128 * GLDA];   // [m][k]
  __shared__ bf16 lds_b[128 * GLDA];   // [n][k]  (B staged transposed = column major)
  const int tid  = threadIdx.x;
  const int lane = tid & 31, wv = tid >> 5;
  const int wm = wv >> 1, wn = wv & 1;
  const int m0 = blockIdx.y * 128, n0 = blockIdx.x * 128;
  const int mrow = lane & 15;
  const int kb16 = (lane >> 4) * 8;

  v8f acc[2][4];
#pragma unroll
  for (int a = 0; a < 2; ++a)
#pragma unroll
    for (int b = 0; b < 4; ++b)
#pragma unroll
      for (int e = 0; e < 8; ++e) acc[a][b][e] = 0.f;

  const int ar = tid >> 1, akb = (tid & 1) * 16;

#if USE_TDM
  // ---- Tensor DMA descriptor (D#), static parts.
  // Tile: 32 elems (dim0, 64B rows) x 128 rows (dim1), data_size=2B.
  // Row = 16 DWORDs; pad_interval=3 (every 16 DWORDs) + pad_amount=3 (4 DWORDs)
  // -> LDS row stride = 20 DWORDs = 40 bf16 = GLDA.
  const unsigned int lds_off =
      (unsigned int)(unsigned long long)(void*)&lds_a[0];
  i32x8 g1;
  g1[0] = (int)((1u << 16) | (1u << 20) | (3u << 22) | (3u << 25));
  g1[1] = (int)(((unsigned)K & 0xFFFFu) << 16);                    // tensor_dim0 lo
  g1[2] = (int)((((unsigned)K >> 16) & 0xFFFFu) |
                (((unsigned)M & 0xFFFFu) << 16));                  // dim0 hi | dim1 lo
  g1[3] = (int)((((unsigned)M >> 16) & 0xFFFFu) | (32u << 16));    // dim1 hi | tile_dim0
  g1[4] = 128;                                                     // tile_dim1
  g1[5] = K;                                                       // dim0_stride lo32
  g1[6] = 0;
  g1[7] = 0;
  const i32x4 gz = {0, 0, 0, 0};
#endif

  for (int k0 = 0; k0 < K; k0 += 32) {
    __syncthreads();
#if USE_TDM
    // ---- A tile 128x32 via Tensor Data Mover (wave 0 issues the DMA)
    if (wv == 0) {
      const unsigned long long ga =
          (unsigned long long)(const void*)(A + (size_t)m0 * K + k0);
      u32x4 g0;
      g0[0] = 1u;                                                  // count = 1
      g0[1] = lds_off;                                             // lds_addr
      g0[2] = (unsigned int)ga;                                    // global_addr lo
      g0[3] = (unsigned int)((ga >> 32) & 0x01FFFFFFu) | (2u << 30); // hi | type=2
#if __clang_major__ >= 23
      const i32x8 gz8 = {0, 0, 0, 0, 0, 0, 0, 0};
      __builtin_amdgcn_tensor_load_to_lds(g0, g1, gz, gz, gz8, 0);
#else
      __builtin_amdgcn_tensor_load_to_lds(g0, g1, gz, gz, 0);
#endif
    }
#else
    // ---- fallback: cooperative A tile load (bf16, 16 elems / thread)
    {
      const v8bf* src = (const v8bf*)(A + (size_t)(m0 + ar) * K + k0 + akb);
      v8bf c0 = src[0], c1 = src[1];
      *(v8bf*)&lds_a[ar * GLDA + akb]     = c0;
      *(v8bf*)&lds_a[ar * GLDA + akb + 8] = c1;
    }
#endif
    // ---- B tile 32x128 : transpose into [n][k] while converting f32->bf16
#pragma unroll
    for (int it = 0; it < 2; ++it) {
      const int c  = tid + it * 256;         // 0..511
      const int kk = c >> 4;                 // 0..31
      const int n  = (c & 15) * 8;           // 0..120
      const float4* src = (const float4*)(Bm + (size_t)(k0 + kk) * N + n0 + n);
      float4 g0 = src[0], g1v = src[1];
      lds_b[(n + 0) * GLDA + kk] = (bf16)g0.x;
      lds_b[(n + 1) * GLDA + kk] = (bf16)g0.y;
      lds_b[(n + 2) * GLDA + kk] = (bf16)g0.z;
      lds_b[(n + 3) * GLDA + kk] = (bf16)g0.w;
      lds_b[(n + 4) * GLDA + kk] = (bf16)g1v.x;
      lds_b[(n + 5) * GLDA + kk] = (bf16)g1v.y;
      lds_b[(n + 6) * GLDA + kk] = (bf16)g1v.z;
      lds_b[(n + 7) * GLDA + kk] = (bf16)g1v.w;
    }
    // ---- prefetch next K-tile of B into cache (global_prefetch_b8)
    if (k0 + 32 < K) {
      __builtin_prefetch(Bm + (size_t)(k0 + 32 + (tid >> 4)) * N + n0 + (tid & 15) * 8, 0, 0);
#if !USE_TDM
      __builtin_prefetch(A + (size_t)(m0 + ar) * K + k0 + 32 + akb, 0, 0);
#endif
    }
#if USE_TDM
    if (wv == 0) __builtin_amdgcn_s_wait_tensorcnt(0);   // DMA done before barrier
#endif
    __syncthreads();
    // ---- WMMA: 2 A-frags x 4 B-frags per wave
    v16bf afr[2];
#pragma unroll
    for (int tm = 0; tm < 2; ++tm)
      afr[tm] = make_frag(&lds_a[(wm * 32 + tm * 16 + mrow) * GLDA + kb16]);
#pragma unroll
    for (int tn = 0; tn < 4; ++tn) {
      v16bf bfr = make_frag(&lds_b[(wn * 64 + tn * 16 + mrow) * GLDA + kb16]);
#pragma unroll
      for (int tm = 0; tm < 2; ++tm)
        acc[tm][tn] = __builtin_amdgcn_wmma_f32_16x16x32_bf16(
            false, afr[tm], false, bfr, (short)0, acc[tm][tn], false, false);
    }
  }
  // ---- epilogue (C/D layout: row = r + 8*(lane>=16), col = lane&15)
#pragma unroll
  for (int tm = 0; tm < 2; ++tm)
#pragma unroll
    for (int tn = 0; tn < 4; ++tn)
#pragma unroll
      for (int r = 0; r < 8; ++r) {
        const int mg = m0 + wm * 32 + tm * 16 + r + 8 * (lane >> 4);
        const int ng = n0 + wn * 64 + tn * 16 + mrow;
        float v = acc[tm][tn][r];
        if (bias) v += bias[ng];
        C[(size_t)mg * N + ng] = v;
      }
}

// ---------------------------------------------------------------------------
// Kernel 3: RoPE on Q,K + split/convert to bf16 in attention layouts.
//  Qf  : [b*s][H*HD]  f32      -> qb [b,H,s,HD]   bf16 (rotated)
//  KVf : [b*s][2*KVH*HD] f32   -> kb [b,KVH,s,HD] bf16 (rotated), vb same (plain)
// ---------------------------------------------------------------------------
__global__ __launch_bounds__(256)
void rope_split_kernel(const float* __restrict__ Qf, const float* __restrict__ KVf,
                       bf16* __restrict__ qb, bf16* __restrict__ kb, bf16* __restrict__ vb) {
  const int bs = blockIdx.x;
  const int b = bs / Sc, s = bs % Sc;
  const int tid = threadIdx.x;
  const float* qr  = Qf  + (size_t)bs * DIMc;
  const float* kvr = KVf + (size_t)bs * DIMc;

  for (int idx = tid; idx < Hc * 64; idx += 256) {
    const int h = idx >> 6, d = idx & 63;
    const float fr = __powf(10000.f, -(float)d * (1.0f / 64.f));
    float sn, cs;
    __sincosf((float)s * fr, &sn, &cs);
    const float x1 = qr[h * HDc + d], x2 = qr[h * HDc + d + 64];
    bf16* qo = qb + (((size_t)b * Hc + h) * Sc + s) * HDc;
    qo[d]      = (bf16)(x1 * cs - x2 * sn);
    qo[d + 64] = (bf16)(x1 * sn + x2 * cs);
  }
  for (int idx = tid; idx < KVHc * 64; idx += 256) {
    const int h = idx >> 6, d = idx & 63;
    const float fr = __powf(10000.f, -(float)d * (1.0f / 64.f));
    float sn, cs;
    __sincosf((float)s * fr, &sn, &cs);
    const float x1 = kvr[h * HDc + d], x2 = kvr[h * HDc + d + 64];
    bf16* ko = kb + (((size_t)b * KVHc + h) * Sc + s) * HDc;
    ko[d]      = (bf16)(x1 * cs - x2 * sn);
    ko[d + 64] = (bf16)(x1 * sn + x2 * cs);
  }
  for (int idx = tid; idx < KVHc * HDc; idx += 256) {
    const int h = idx >> 7, d = idx & 127;
    vb[(((size_t)b * KVHc + h) * Sc + s) * HDc + d] =
        (bf16)kvr[KVHc * HDc + h * HDc + d];
  }
}

// ---------------------------------------------------------------------------
// Kernel 4: flash-style attention core, bf16 WMMA for QK^T and P*V.
// Block = 64 query rows of one (b,h); 8 wave32 waves; K/V streamed in 64-token
// tiles through LDS.  ALiBi + inverted-window mask applied in f32 epilogue.
// Output written directly as bf16 [b,s,h*HD] (A-operand of the final GEMM).
// ---------------------------------------------------------------------------
__global__ __launch_bounds__(256)
void attn_kernel(const bf16* __restrict__ Q, const bf16* __restrict__ Kt,
                 const bf16* __restrict__ Vt, bf16* __restrict__ O) {
  constexpr int LDQ = 136;   // 128 + 8 pad (bf16 elems)
  constexpr int LDV = 72;    // 64 + 8 pad
  constexpr int LDSS = 72;
  __shared__ bf16  q_s[64 * LDQ];    // [token][d]
  __shared__ bf16  k_s[64 * LDQ];    // [token][d]
  __shared__ bf16  v_s[128 * LDV];   // transposed: [d][token]
  __shared__ bf16  p_s[64 * LDSS];   // softmax probs (bf16) [i][j]
  __shared__ float s_s[64 * LDSS];   // raw scores f32 [i][j]
  __shared__ float m_row[64], l_row[64], a_row[64];

  const int qt = blockIdx.x, h = blockIdx.y, b = blockIdx.z;
  const int kvh = h >> 1;                       // GQA repeat factor 2
  const int tid = threadIdx.x, lane = tid & 31, wv = tid >> 5;
  const int q0 = qt * 64;
  const int mrow = lane & 15, kb16 = (lane >> 4) * 8;
  const int om = wv >> 1, onb = (wv & 1) * 4;   // O-tile ownership (4x8 grid)

  const bf16* qg = Q  + ((size_t)b * Hc   + h)   * Sc * HDc;
  const bf16* kg = Kt + ((size_t)b * KVHc + kvh) * Sc * HDc;
  const bf16* vg = Vt + ((size_t)b * KVHc + kvh) * Sc * HDc;

  {   // load 64x128 Q tile
    const int r = tid >> 2, db = (tid & 3) * 32;
    const v8bf* src = (const v8bf*)(qg + (size_t)(q0 + r) * HDc + db);
    v8bf t0 = src[0], t1 = src[1], t2 = src[2], t3 = src[3];
    v8bf* dst = (v8bf*)&q_s[r * LDQ + db];
    dst[0] = t0; dst[1] = t1; dst[2] = t2; dst[3] = t3;
  }
  if (tid < 64) { m_row[tid] = -3.0e38f; l_row[tid] = 0.f; }

  const float slope = exp2f(-0.5f * (float)(h + 1));   // ALiBi slope for H=16

  v8f oacc[4];
#pragma unroll
  for (int t = 0; t < 4; ++t)
#pragma unroll
    for (int e = 0; e < 8; ++e) oacc[t][e] = 0.f;

  for (int j0 = 0; j0 < Sc; j0 += 64) {
    __syncthreads();   // previous iteration done with k_s/v_s/p_s
    {   // load K tile row-major, V tile transposed
      const int r = tid >> 2, db = (tid & 3) * 32;
      const v8bf* ks = (const v8bf*)(kg + (size_t)(j0 + r) * HDc + db);
      v8bf a0 = ks[0], a1 = ks[1], a2 = ks[2], a3 = ks[3];
      v8bf* kd = (v8bf*)&k_s[r * LDQ + db];
      kd[0] = a0; kd[1] = a1; kd[2] = a2; kd[3] = a3;
      const v8bf* vs = (const v8bf*)(vg + (size_t)(j0 + r) * HDc + db);
      v8bf b0 = vs[0], b1 = vs[1], b2 = vs[2], b3 = vs[3];
#pragma unroll
      for (int i = 0; i < 8; ++i) {
        v_s[(db + i)      * LDV + r] = b0[i];
        v_s[(db + 8 + i)  * LDV + r] = b1[i];
        v_s[(db + 16 + i) * LDV + r] = b2[i];
        v_s[(db + 24 + i) * LDV + r] = b3[i];
      }
    }
    __syncthreads();
    // ---- S = Q K^T : 16 tiles of 16x16, 2 per wave, K-loop over HD=128
#pragma unroll
    for (int it = 0; it < 2; ++it) {
      const int id = wv * 2 + it;
      const int tm = id >> 2, tn = id & 3;
      v8f acc;
#pragma unroll
      for (int e = 0; e < 8; ++e) acc[e] = 0.f;
#pragma unroll
      for (int kk = 0; kk < HDc; kk += 32) {
        v16bf af = make_frag(&q_s[(tm * 16 + mrow) * LDQ + kk + kb16]);
        v16bf bf = make_frag(&k_s[(tn * 16 + mrow) * LDQ + kk + kb16]);
        acc = __builtin_amdgcn_wmma_f32_16x16x32_bf16(false, af, false, bf,
                                                      (short)0, acc, false, false);
      }
#pragma unroll
      for (int r = 0; r < 8; ++r) {
        const int il = tm * 16 + r + 8 * (lane >> 4);
        const int jl = tn * 16 + mrow;
        const int ig = q0 + il, jg = j0 + jl;
        const int dist = ig > jg ? ig - jg : jg - ig;
        // faithful inverted mask: keep qk only OUTSIDE window & non-global
        const bool keep = (dist > 256) && (ig >= 64) && (jg >= 64);
        const float v = acc[r] * SCALEc - slope * (float)dist;
        s_s[il * LDSS + jl] = keep ? v : NEGc;
      }
    }
    __syncthreads();
    // ---- online softmax (one thread per row)
    if (tid < 64) {
      const float* sr = &s_s[tid * LDSS];
      const float mo = m_row[tid];
      float mn = mo;
#pragma unroll 8
      for (int c = 0; c < 64; ++c) mn = fmaxf(mn, sr[c]);
      const float al = __expf(mo - mn);
      float ls = 0.f;
      bf16* pr = &p_s[tid * LDSS];
#pragma unroll 8
      for (int c = 0; c < 64; ++c) {
        const float pv = __expf(sr[c] - mn);
        ls += pv;
        pr[c] = (bf16)pv;
      }
      m_row[tid] = mn;
      l_row[tid] = l_row[tid] * al + ls;
      a_row[tid] = al;
    }
    __syncthreads();
    // ---- rescale persistent O accumulators, then O += P V
    float al[8];
#pragma unroll
    for (int r = 0; r < 8; ++r) al[r] = a_row[om * 16 + r + 8 * (lane >> 4)];
#pragma unroll
    for (int t = 0; t < 4; ++t)
#pragma unroll
      for (int r = 0; r < 8; ++r) oacc[t][r] *= al[r];
#pragma unroll
    for (int kk = 0; kk < 64; kk += 32) {
      v16bf af = make_frag(&p_s[(om * 16 + mrow) * LDSS + kk + kb16]);
#pragma unroll
      for (int t = 0; t < 4; ++t) {
        const int on = onb + t;
        v16bf bf = make_frag(&v_s[(on * 16 + mrow) * LDV + kk + kb16]);
        oacc[t] = __builtin_amdgcn_wmma_f32_16x16x32_bf16(false, af, false, bf,
                                                          (short)0, oacc[t], false, false);
      }
    }
  }
  // ---- normalize and store bf16 [b, s, h*HD + d] (A operand of final GEMM)
#pragma unroll
  for (int t = 0; t < 4; ++t) {
    const int on = onb + t;
#pragma unroll
    for (int r = 0; r < 8; ++r) {
      const int il = om * 16 + r + 8 * (lane >> 4);
      const int d  = on * 16 + mrow;
      const float inv = 1.0f / l_row[il];
      O[((size_t)b * Sc + (q0 + il)) * DIMc + h * HDc + d] = (bf16)(oacc[t][r] * inv);
    }
  }
}

// ---------------------------------------------------------------------------
// Host-side launcher
// ---------------------------------------------------------------------------
extern "C" void kernel_launch(void* const* d_in, const int* in_sizes, int n_in,
                              void* d_out, int out_size, void* d_ws, size_t ws_size,
                              hipStream_t stream) {
  (void)in_sizes; (void)n_in; (void)out_size; (void)ws_size;
  const float* x   = (const float*)d_in[0];
  const float* Wq  = (const float*)d_in[1];
  const float* Wkv = (const float*)d_in[2];
  const float* Wo  = (const float*)d_in[3];
  const float* bo  = (const float*)d_in[4];
  float* out = (float*)d_out;

  const size_t MR = (size_t)Bc * Sc;   // 4096 rows
  char* p = (char*)d_ws;
  bf16*  xgb = (bf16*)p;  p += MR * DIMc * sizeof(bf16);   // gated x  (bf16)
  bf16*  xb  = (bf16*)p;  p += MR * DIMc * sizeof(bf16);   // plain x  (bf16)
  float* Qf  = (float*)p; p += MR * DIMc * sizeof(float);
  float* KVf = (float*)p; p += MR * DIMc * sizeof(float);
  bf16* qb = (bf16*)p; p += (size_t)Bc * Hc   * Sc * HDc * sizeof(bf16);
  bf16* kb = (bf16*)p; p += (size_t)Bc * KVHc * Sc * HDc * sizeof(bf16);
  bf16* vb = (bf16*)p; p += (size_t)Bc * KVHc * Sc * HDc * sizeof(bf16);
  bf16* attnb = xgb;   // xgb is dead after the Q projection -> reuse

  gate_kernel<<<dim3((unsigned)MR), 256, 0, stream>>>(x, xgb);
  {
    const long long n8 = (long long)(MR * DIMc / 8);
    f32_to_bf16_kernel<<<dim3((unsigned)((n8 + 255) / 256)), 256, 0, stream>>>(x, xb, n8);
  }

  dim3 ggrid(DIMc / 128, (unsigned)(MR / 128));
  gemm_bf16_wmma_kernel<<<ggrid, 256, 0, stream>>>(xgb, Wq,  nullptr, Qf,  (int)MR, DIMc, DIMc);
  gemm_bf16_wmma_kernel<<<ggrid, 256, 0, stream>>>(xb,  Wkv, nullptr, KVf, (int)MR, DIMc, DIMc);

  rope_split_kernel<<<dim3((unsigned)MR), 256, 0, stream>>>(Qf, KVf, qb, kb, vb);

  attn_kernel<<<dim3(Sc / 64, Hc, Bc), 256, 0, stream>>>(qb, kb, vb, attnb);

  gemm_bf16_wmma_kernel<<<ggrid, 256, 0, stream>>>(attnb, Wo, bo, out, (int)MR, DIMc, DIMc);
}
